// WalkFormer_51470888075606
// MI455X (gfx1250) — compile-verified, matrix operand order checked by reading, not run on hardware
//
#include <hip/hip_runtime.h>
#include <hip/hip_bf16.h>

// ---------------- problem constants ----------------
constexpr int Bc = 4, Nc = 16384, Mc = 512, Cc = 128, Kc = 16, Lc = 6;
constexpr int STEPSc = 5, Hc = 4, DHc = Cc / Hc;          // 4 heads, Dh=32
constexpr int Rpts = Bc * Nc;                             // 65536 encoder rows
constexpr int Rseq = Bc * Mc * Lc;                        // 12288 route rows
constexpr float INV_SQRT_C  = 0.08838834764831843f;       // 1/sqrt(128)
constexpr float INV_SQRT_DH = 0.17677669529663687f;       // 1/sqrt(32)
#define KK 17                                             // K+1 nearest

typedef __attribute__((ext_vector_type(16))) __bf16 v16bf;
typedef __attribute__((ext_vector_type(8)))  float  v8f;

// ---------------- helpers ----------------
static __device__ __forceinline__ float wave_sum(float v) {
#pragma unroll
  for (int o = 16; o > 0; o >>= 1) v += __shfl_xor(v, o, 32);
  return v;
}

// ---------------- f32 -> bf16 convert ----------------
__global__ void cvt_bf16_kernel(const float* __restrict__ src, __bf16* __restrict__ dst, int n) {
  int i = blockIdx.x * blockDim.x + threadIdx.x;
  if (i < n) dst[i] = (__bf16)src[i];
}

// ---------------- encoder: xyz -> relu(3->64) -> 128, store bf16 ----------------
__global__ void encoder_kernel(const float* __restrict__ xyz,
                               const float* __restrict__ w1, const float* __restrict__ b1,
                               const float* __restrict__ w2, const float* __restrict__ b2,
                               __bf16* __restrict__ feats_bf) {
  int p = blockIdx.x * blockDim.x + threadIdx.x;
  if (p >= Rpts) return;
  float x0 = xyz[p * 3 + 0], x1 = xyz[p * 3 + 1], x2 = xyz[p * 3 + 2];
  float hid[64];
#pragma unroll
  for (int h = 0; h < 64; ++h) {
    float s = b1[h] + x0 * w1[h * 3 + 0] + x1 * w1[h * 3 + 1] + x2 * w1[h * 3 + 2];
    hid[h] = fmaxf(s, 0.0f);
  }
  for (int c = 0; c < Cc; ++c) {
    float acc = b2[c];
#pragma unroll
    for (int h = 0; h < 64; ++h) acc += hid[h] * w2[c * 64 + h];
    feats_bf[(size_t)p * Cc + c] = (__bf16)acc;
  }
}

// ---------------- generic bf16 WMMA GEMM: out[r][c] = act(A[r,:] . W[c,:] + bias[c]) ----------
// A: R x Kd (bf16 row-major), W: Cout x Kd (bf16 row-major, i.e. B = W^T)
// grid.x = R/16, grid.y = Cout/128, block = 256 (8 waves, one 16x16 tile each)
__global__ void gemm_bf16_wmma_kernel(const __bf16* __restrict__ A,
                                      const __bf16* __restrict__ W,
                                      const float*  __restrict__ bias,
                                      float*  __restrict__ outF,
                                      __bf16* __restrict__ outB,
                                      int Kd, int Cout, int relu) {
  const int lane   = threadIdx.x & 31;
  const int wave   = threadIdx.x >> 5;
  const int rowBase = blockIdx.x * 16;
  const int colBase = (blockIdx.y * 8 + wave) * 16;
  const int rhalf  = lane >> 4;     // K sub-group select
  const int rl     = lane & 15;     // A row / B col within tile

  const __bf16* arow = A + (size_t)(rowBase + rl) * Kd + rhalf * 8;
  const __bf16* wrow = W + (size_t)(colBase + rl) * Kd + rhalf * 8;

  v8f acc;
  float bv = bias ? bias[colBase + rl] : 0.0f;
#pragma unroll
  for (int i = 0; i < 8; ++i) acc[i] = bv;

  for (int kc = 0; kc < Kd; kc += 32) {
    v16bf a, b;
    ((uint4*)&a)[0] = *(const uint4*)(arow + kc);
    ((uint4*)&a)[1] = *(const uint4*)(arow + kc + 16);
    ((uint4*)&b)[0] = *(const uint4*)(wrow + kc);
    ((uint4*)&b)[1] = *(const uint4*)(wrow + kc + 16);
    __builtin_prefetch(arow + kc + 64, 0, 1);
    acc = __builtin_amdgcn_wmma_f32_16x16x32_bf16(false, a, false, b, (short)0, acc,
                                                  false, false);
  }

  const int orow0 = rowBase + rhalf * 8;
  const int ocol  = colBase + rl;
#pragma unroll
  for (int i = 0; i < 8; ++i) {
    float v = acc[i];
    if (relu) v = fmaxf(v, 0.0f);
    size_t off = (size_t)(orow0 + i) * Cout + ocol;
    if (outF) outF[off] = v;
    if (outB) outB[off] = (__bf16)v;
  }
}

// ---------------- per-step init: cur_idx / walk_idx[...,0] = start ----------------
__global__ void init_idx_kernel(const int* __restrict__ start,
                                int* __restrict__ cur_idx, int* __restrict__ walk_idx) {
  int i = blockIdx.x * blockDim.x + threadIdx.x;
  if (i < Bc * Mc) { cur_idx[i] = start[i]; walk_idx[i * Lc] = start[i]; }
}

// ---------------- KNN: wave per walker, top-17 smallest d2 over N, drop nearest ---------
__global__ void knn_kernel(const float* __restrict__ pos, const int* __restrict__ cur_idx,
                           int* __restrict__ knn) {
  __shared__ float sd[8][KK * 32];
  __shared__ int   si[8][KK * 32];
  int lane = threadIdx.x & 31, wave = threadIdx.x >> 5;
  int wid = blockIdx.x * 8 + wave;                 // b*M+m
  int b = wid / Mc;
  const float* pb = pos + (size_t)b * Nc * 3;
  int ci = cur_idx[wid];
  float px = pb[ci * 3 + 0], py = pb[ci * 3 + 1], pz = pb[ci * 3 + 2];

  float bd[KK]; int bi[KK];
#pragma unroll
  for (int s = 0; s < KK; ++s) { bd[s] = 3.4e38f; bi[s] = 0x7fffffff; }

  for (int n = lane; n < Nc; n += 32) {
    float dx = pb[n * 3 + 0] - px, dy = pb[n * 3 + 1] - py, dz = pb[n * 3 + 2] - pz;
    float d2 = dx * dx + dy * dy + dz * dz;
    if (d2 < bd[KK - 1]) {                         // register insertion network
      bool up = true;
#pragma unroll
      for (int s = KK - 1; s > 0; --s) {
        bool sh = d2 < bd[s - 1];
        float nv = sh ? bd[s - 1] : (up ? d2 : bd[s]);
        int   ni = sh ? bi[s - 1] : (up ? n  : bi[s]);
        bd[s] = nv; bi[s] = ni; up = sh;
      }
      if (up) { bd[0] = d2; bi[0] = n; }
    }
  }

#pragma unroll
  for (int s = 0; s < KK; ++s) {
    sd[wave][lane * KK + s] = bd[s];
    si[wave][lane * KK + s] = bi[s];
  }
  __syncthreads();

  int head = 0;                                    // 17 rounds of wave-wide min merge
  for (int r = 0; r < KK; ++r) {
    float d  = (head < KK) ? sd[wave][lane * KK + head] : 3.4e38f;
    int   id = (head < KK) ? si[wave][lane * KK + head] : 0x7fffffff;
    int   ln = lane;
#pragma unroll
    for (int o = 16; o > 0; o >>= 1) {
      float od = __shfl_xor(d, o, 32);
      int   oi = __shfl_xor(id, o, 32);
      int   ol = __shfl_xor(ln, o, 32);
      if (od < d || (od == d && oi < id)) { d = od; id = oi; ln = ol; }
    }
    if (lane == ln) head++;
    if (r > 0 && lane == 0) knn[wid * Kc + (r - 1)] = id;  // drop nearest (self)
  }
}

// ---------------- choice: logits = bfeat[i].gfeat[knn]/sqrt(C) + gumbel; argmax ---------
__global__ void choice_kernel(const __bf16* __restrict__ bfeat, const __bf16* __restrict__ gfeat,
                              const int* __restrict__ knn, const float* __restrict__ gum,
                              int* __restrict__ cur_idx, int* __restrict__ walk_idx, int l) {
  int lane = threadIdx.x & 31, wave = threadIdx.x >> 5;
  int wid = blockIdx.x * 8 + wave;
  int b = wid / Mc;
  int ci = cur_idx[wid];
  const __bf16* qrow = bfeat + ((size_t)b * Nc + ci) * Cc;
  float qv[4];
#pragma unroll
  for (int j = 0; j < 4; ++j) qv[j] = (float)qrow[lane * 4 + j];

  float myscore = -3.4e38f;
  for (int k = 0; k < Kc; ++k) {
    int gi = knn[wid * Kc + k];
    const __bf16* grow = gfeat + ((size_t)b * Nc + gi) * Cc;
    float p = 0.0f;
#pragma unroll
    for (int j = 0; j < 4; ++j) p += qv[j] * (float)grow[lane * 4 + j];
    p = wave_sum(p);
    if (lane == k) myscore = p * INV_SQRT_C + gum[wid * Kc + k];
  }
  // argmax over lanes 0..15 (ties -> lowest k, matching jnp.argmax)
  float s = (lane < Kc) ? myscore : -3.4e38f;
  int   kk = (lane < Kc) ? lane : Kc;
#pragma unroll
  for (int o = 16; o > 0; o >>= 1) {
    float os = __shfl_xor(s, o, 32);
    int   ok = __shfl_xor(kk, o, 32);
    if (os > s || (os == s && ok < kk)) { s = os; kk = ok; }
  }
  int ni = knn[wid * Kc + kk];
  if (lane == 0) { cur_idx[wid] = ni; walk_idx[wid * Lc + (l + 1)] = ni; }
}

// ---------------- gather walk features (bf16 rows, 256B per row) ----------------
__global__ void gather_walk_kernel(const __bf16* __restrict__ feats_bf,
                                   const int* __restrict__ walk_idx,
                                   __bf16* __restrict__ curr_bf) {
  int lane = threadIdx.x & 31, wave = threadIdx.x >> 5;
  int row = blockIdx.x * 8 + wave;                 // (b*M+m)*L + l
  int bm  = row / Lc;
  int b   = bm / Mc;
  int idx = walk_idx[row];
  const uint2* src = (const uint2*)(feats_bf + ((size_t)b * Nc + idx) * Cc);
  uint2* dst = (uint2*)(curr_bf + (size_t)row * Cc);
  dst[lane] = src[lane];
}

// ---------------- attention: wave per (sequence, head), L=6, Dh=32 ----------------
__global__ void attn_kernel(const float* __restrict__ qbuf, const float* __restrict__ kvbuf,
                            __bf16* __restrict__ attn_o) {
  int lane = threadIdx.x & 31, wave = threadIdx.x >> 5;
  int wid = blockIdx.x * 8 + wave;                 // seq*H + h
  int seq = wid / Hc, h = wid % Hc;
  float qv[Lc], kv[Lc], vv[Lc];
#pragma unroll
  for (int i = 0; i < Lc; ++i) {
    size_t r = (size_t)seq * Lc + i;
    qv[i] = qbuf [r * Cc       + h * DHc + lane];
    kv[i] = kvbuf[r * (2 * Cc) + h * DHc + lane];
    vv[i] = kvbuf[r * (2 * Cc) + Cc + h * DHc + lane];
  }
  float S[Lc][Lc];
#pragma unroll
  for (int i = 0; i < Lc; ++i)
#pragma unroll
    for (int j = 0; j < Lc; ++j)
      S[i][j] = wave_sum(qv[i] * kv[j]) * INV_SQRT_DH;

#pragma unroll
  for (int i = 0; i < Lc; ++i) {
    float mx = S[i][0];
#pragma unroll
    for (int j = 1; j < Lc; ++j) mx = fmaxf(mx, S[i][j]);
    float den = 0.0f;
#pragma unroll
    for (int j = 0; j < Lc; ++j) { S[i][j] = expf(S[i][j] - mx); den += S[i][j]; }
    float inv = 1.0f / den;
    float o = 0.0f;
#pragma unroll
    for (int j = 0; j < Lc; ++j) o += S[i][j] * inv * vv[j];
    attn_o[((size_t)seq * Lc + i) * Cc + h * DHc + lane] = (__bf16)o;
  }
}

// ---------------- predict: 256 -> relu 64 -> 3 -> tanh -> scatter-add ----------------
__global__ void predict_kernel(const float* __restrict__ routed, const int* __restrict__ start,
                               const float* __restrict__ w1, const float* __restrict__ b1,
                               const float* __restrict__ w2, const float* __restrict__ b2,
                               float* __restrict__ pos) {
  __shared__ float xs[8][2 * Cc];
  __shared__ float hs[8][64];
  int lane = threadIdx.x & 31, wave = threadIdx.x >> 5;
  int wid = blockIdx.x * 8 + wave;                 // b*M+m
  int b = wid / Mc;
  const float* wr = routed + (size_t)wid * Lc * Cc;
#pragma unroll
  for (int j = 0; j < 4; ++j) {
    int c = lane * 4 + j;
    float cen = wr[c];
    float s = 0.0f;
#pragma unroll
    for (int l = 1; l < Lc; ++l) s += wr[l * Cc + c];
    xs[wave][c]      = s * (1.0f / (Lc - 1)) - cen; // mean of (walk[l]-central)
    xs[wave][Cc + c] = cen;
  }
  __syncthreads();
#pragma unroll
  for (int j = 0; j < 2; ++j) {
    int hh = lane * 2 + j;
    float acc = b1[hh];
    for (int c = 0; c < 2 * Cc; ++c) acc += xs[wave][c] * w1[hh * 2 * Cc + c];
    hs[wave][hh] = fmaxf(acc, 0.0f);
  }
  __syncthreads();
  if (lane < 3) {
    float acc = b2[lane];
#pragma unroll
    for (int c = 0; c < 64; ++c) acc += hs[wave][c] * w2[lane * 64 + c];
    float disp = tanhf(acc);
    int s0 = start[wid];
    atomicAdd(&pos[((size_t)b * Nc + s0) * 3 + lane], disp);
  }
}

// ---------------- host orchestration ----------------
extern "C" void kernel_launch(void* const* d_in, const int* in_sizes, int n_in,
                              void* d_out, int out_size, void* d_ws, size_t ws_size,
                              hipStream_t stream) {
  (void)in_sizes; (void)n_in; (void)out_size; (void)ws_size;
  const float* xyz       = (const float*)d_in[0];
  const int*   start     = (const int*)  d_in[1];
  const float* gumbel    = (const float*)d_in[2];
  const float* enc_w1    = (const float*)d_in[3];
  const float* enc_b1    = (const float*)d_in[4];
  const float* enc_w2    = (const float*)d_in[5];
  const float* enc_b2    = (const float*)d_in[6];
  const float* beta_w    = (const float*)d_in[7];
  const float* beta_b    = (const float*)d_in[8];
  const float* gamma_w   = (const float*)d_in[9];
  const float* gamma_b   = (const float*)d_in[10];
  const float* attn_w_in = (const float*)d_in[11];
  const float* attn_b_in = (const float*)d_in[12];
  const float* attn_w_out= (const float*)d_in[13];
  const float* attn_b_out= (const float*)d_in[14];
  const float* rt_w1     = (const float*)d_in[15];
  const float* rt_b1     = (const float*)d_in[16];
  const float* rt_w2     = (const float*)d_in[17];
  const float* rt_b2     = (const float*)d_in[18];
  const float* pred_w1   = (const float*)d_in[19];
  const float* pred_b1   = (const float*)d_in[20];
  const float* pred_w2   = (const float*)d_in[21];
  const float* pred_b2   = (const float*)d_in[22];

  float* pos = (float*)d_out;                      // working positions (B,N,3)

  char* ws = (char*)d_ws;
  size_t off = 0;
  auto alloc = [&](size_t bytes) -> void* {
    void* p = ws + off;
    off = (off + bytes + 255) & ~(size_t)255;
    return p;
  };
  __bf16* feats_bf  = (__bf16*)alloc((size_t)Rpts * Cc * 2);
  __bf16* bfeat_bf  = (__bf16*)alloc((size_t)Rpts * Cc * 2);
  __bf16* gfeat_bf  = (__bf16*)alloc((size_t)Rpts * Cc * 2);
  __bf16* wbf       = (__bf16*)alloc((size_t)131072 * 2);  // packed bf16 weights
  int*    cur_idx   = (int*)   alloc((size_t)Bc * Mc * 4);
  int*    walk_idx  = (int*)   alloc((size_t)Bc * Mc * Lc * 4);
  int*    knn_buf   = (int*)   alloc((size_t)Bc * Mc * Kc * 4);
  __bf16* curr_bf   = (__bf16*)alloc((size_t)Rseq * Cc * 2);
  __bf16* routed_bf = (__bf16*)alloc((size_t)Rseq * Cc * 2);
  float*  qbuf      = (float*) alloc((size_t)Rseq * Cc * 4);
  float*  kvbuf     = (float*) alloc((size_t)Rseq * 2 * Cc * 4);
  __bf16* attn_o_bf = (__bf16*)alloc((size_t)Rseq * Cc * 2);
  __bf16* o1_bf     = (__bf16*)alloc((size_t)Rseq * Cc * 2);
  __bf16* h_bf      = (__bf16*)alloc((size_t)Rseq * Cc * 2);
  float*  routed_f  = (float*) alloc((size_t)Rseq * Cc * 4);

  __bf16* w_beta = wbf + 0;
  __bf16* w_gamma= wbf + 16384;
  __bf16* w_qkv  = wbf + 32768;     // 384x128
  __bf16* w_out  = wbf + 81920;
  __bf16* w_rt1  = wbf + 98304;
  __bf16* w_rt2  = wbf + 114688;

  // positions start as xyz
  hipMemcpyAsync(pos, xyz, (size_t)Bc * Nc * 3 * 4, hipMemcpyDeviceToDevice, stream);

  // pack weights to bf16
  cvt_bf16_kernel<<<(16384 + 255) / 256, 256, 0, stream>>>(beta_w,    w_beta, 16384);
  cvt_bf16_kernel<<<(16384 + 255) / 256, 256, 0, stream>>>(gamma_w,   w_gamma,16384);
  cvt_bf16_kernel<<<(49152 + 255) / 256, 256, 0, stream>>>(attn_w_in, w_qkv,  49152);
  cvt_bf16_kernel<<<(16384 + 255) / 256, 256, 0, stream>>>(attn_w_out,w_out,  16384);
  cvt_bf16_kernel<<<(16384 + 255) / 256, 256, 0, stream>>>(rt_w1,     w_rt1,  16384);
  cvt_bf16_kernel<<<(16384 + 255) / 256, 256, 0, stream>>>(rt_w2,     w_rt2,  16384);

  // encoder
  encoder_kernel<<<Rpts / 256, 256, 0, stream>>>(xyz, enc_w1, enc_b1, enc_w2, enc_b2, feats_bf);

  // one-time beta/gamma projections (hoisted out of the 25 selector rounds)
  gemm_bf16_wmma_kernel<<<dim3(Rpts / 16, 1), 256, 0, stream>>>(
      feats_bf, w_beta, beta_b, nullptr, bfeat_bf, Cc, Cc, 0);
  gemm_bf16_wmma_kernel<<<dim3(Rpts / 16, 1), 256, 0, stream>>>(
      feats_bf, w_gamma, gamma_b, nullptr, gfeat_bf, Cc, Cc, 0);

  for (int t = 0; t < STEPSc; ++t) {
    init_idx_kernel<<<(Bc * Mc + 255) / 256, 256, 0, stream>>>(start, cur_idx, walk_idx);
    for (int l = 0; l < Lc - 1; ++l) {
      knn_kernel<<<Bc * Mc / 8, 256, 0, stream>>>(pos, cur_idx, knn_buf);
      const float* gum = gumbel + (size_t)(t * (Lc - 1) + l) * Bc * Mc * Kc;
      choice_kernel<<<Bc * Mc / 8, 256, 0, stream>>>(bfeat_bf, gfeat_bf, knn_buf, gum,
                                                     cur_idx, walk_idx, l);
    }
    gather_walk_kernel<<<Rseq / 8, 256, 0, stream>>>(feats_bf, walk_idx, curr_bf);

    const __bf16* prev_bf = (t == 0) ? curr_bf : routed_bf;
    // q from curr, k|v from prev (rows 128..383 of attn_w_in)
    gemm_bf16_wmma_kernel<<<dim3(Rseq / 16, 1), 256, 0, stream>>>(
        curr_bf, w_qkv, attn_b_in, qbuf, nullptr, Cc, Cc, 0);
    gemm_bf16_wmma_kernel<<<dim3(Rseq / 16, 2), 256, 0, stream>>>(
        prev_bf, w_qkv + (size_t)Cc * Cc, attn_b_in + Cc, kvbuf, nullptr, Cc, 2 * Cc, 0);

    attn_kernel<<<Bc * Mc * Hc / 8, 256, 0, stream>>>(qbuf, kvbuf, attn_o_bf);

    gemm_bf16_wmma_kernel<<<dim3(Rseq / 16, 1), 256, 0, stream>>>(
        attn_o_bf, w_out, attn_b_out, nullptr, o1_bf, Cc, Cc, 0);
    gemm_bf16_wmma_kernel<<<dim3(Rseq / 16, 1), 256, 0, stream>>>(
        o1_bf, w_rt1, rt_b1, nullptr, h_bf, Cc, Cc, 1);           // fused ReLU
    gemm_bf16_wmma_kernel<<<dim3(Rseq / 16, 1), 256, 0, stream>>>(
        h_bf, w_rt2, rt_b2, routed_f, routed_bf, Cc, Cc, 0);      // f32 for predict, bf16 for next prev

    predict_kernel<<<Bc * Mc / 8, 256, 0, stream>>>(routed_f, start,
                                                    pred_w1, pred_b1, pred_w2, pred_b2, pos);
  }
}
